// EdgeProcessor_2147483648135
// MI455X (gfx1250) — compile-verified, matrix operand order checked by reading, not run on hardware
//
#include <hip/hip_runtime.h>
#include <hip/hip_bf16.h>
#include <math.h>

// Problem constants (match reference)
#define E_CNT    262144
#define EDIM     128
#define H2_DIM   256
#define T_CNT    8
#define LN_EPS   1e-5f

typedef __attribute__((ext_vector_type(16))) _Float16 v16h;
typedef __attribute__((ext_vector_type(8)))  _Float16 v8h;
typedef __attribute__((ext_vector_type(8)))  float    v8f;
typedef __attribute__((ext_vector_type(4)))  float    v4f;

// Workspace layout (int units):
//   [0..7]        counts[t]
//   [8..15]       cursor[t]      (scatter cursors)
//   [16..24]      base[t]        (exclusive prefix of counts, 9 entries)
//   [32..40]      tileStart[t]   (exclusive prefix of ceil(counts/16), 9 entries)
//   [64..64+E)    idxbuf         (edge ids bucketed by type)
//   then (16B aligned): Wt f16 [T][H2][EDIM]  (transposed, K contiguous)
// Total ~1.6 MB.

__global__ void k_zero(int* ws) {
  if (threadIdx.x < 16) ws[threadIdx.x] = 0;
}

__global__ void k_hist(const int* __restrict__ etype, int* __restrict__ counts) {
  int e = blockIdx.x * blockDim.x + threadIdx.x;
  if (e < E_CNT) atomicAdd(&counts[etype[e] & (T_CNT - 1)], 1);
}

__global__ void k_prefix(int* ws) {
  if (threadIdx.x == 0) {
    int* counts = ws; int* cursor = ws + 8; int* base = ws + 16; int* ts = ws + 32;
    base[0] = 0; ts[0] = 0;
    for (int t = 0; t < T_CNT; ++t) {
      base[t + 1] = base[t] + counts[t];
      ts[t + 1]   = ts[t] + (counts[t] + 15) / 16;
      cursor[t]   = base[t];
    }
  }
}

__global__ void k_scatter(const int* __restrict__ etype, int* __restrict__ cursor,
                          int* __restrict__ idxbuf) {
  int e = blockIdx.x * blockDim.x + threadIdx.x;
  if (e < E_CNT) {
    int t = etype[e] & (T_CNT - 1);
    int p = atomicAdd(&cursor[t], 1);
    idxbuf[p] = e;
  }
}

// W [T][K=128][N=256] f32  ->  Wt [T][N=256][K=128] f16 (K contiguous)
__global__ void k_convW(const float* __restrict__ W, _Float16* __restrict__ Wt) {
  int i = blockIdx.x * blockDim.x + threadIdx.x;
  if (i < T_CNT * H2_DIM * EDIM) {
    int t = i / (H2_DIM * EDIM);
    int rem = i - t * (H2_DIM * EDIM);
    int n = rem / EDIM;
    int k = rem - n * EDIM;
    Wt[i] = (_Float16)W[((size_t)t * EDIM + k) * H2_DIM + n];
  }
}

// One wave32 per 16-row tile of one expert: 16x128 (A,f16) x 128x256 (B,f16)
// -> 16x256 f32, then fused bias + LayerNorm + exact-erf GELU, scatter-store.
__global__ __launch_bounds__(32) void k_moe(
    const float* __restrict__ X,  const float* __restrict__ Bv,
    const float* __restrict__ G,  const float* __restrict__ Be,
    const int*  __restrict__ ws,  const _Float16* __restrict__ Wt,
    float* __restrict__ Y)
{
  const int* counts = ws;
  const int* base   = ws + 16;
  const int* ts     = ws + 32;
  const int* idxbuf = ws + 64;

  int tile = blockIdx.x;
  if (tile >= ts[T_CNT]) return;          // uniform exit (EXEC all-1 for WMMA)

  int t = 0;
  #pragma unroll
  for (int u = 0; u < T_CNT - 1; ++u) if (tile >= ts[u + 1]) t = u + 1;

  int rbase  = base[t] + (tile - ts[t]) * 16;
  int rcount = min(16, base[t] + counts[t] - rbase);

  int lane = threadIdx.x;
  int low  = lane & 15;
  int hi   = lane >> 4;

  // ---- A fragments: row M = low, K-chunks {0..7,16..23} (hi=0) / {8..15,24..31} (hi=1)
  int am = (low < rcount) ? low : 0;      // clamp ragged tile (valid dummy row)
  const float* arow = X + (size_t)idxbuf[rbase + am] * EDIM;
  v16h a[4];
  #pragma unroll
  for (int ks = 0; ks < 4; ++ks) {
    int k0 = 32 * ks + hi * 8;
    v4f f0 = *(const v4f*)(arow + k0);
    v4f f1 = *(const v4f*)(arow + k0 + 4);
    v4f f2 = *(const v4f*)(arow + k0 + 16);
    v4f f3 = *(const v4f*)(arow + k0 + 20);
    v16h av;
    #pragma unroll
    for (int i = 0; i < 4; ++i) {
      av[i]      = (_Float16)f0[i];
      av[4 + i]  = (_Float16)f1[i];
      av[8 + i]  = (_Float16)f2[i];
      av[12 + i] = (_Float16)f3[i];
    }
    a[ks] = av;
  }

  // ---- GEMM: 16 N-tiles x 4 K-steps of v_wmma_f32_16x16x32_f16
  const _Float16* wt = Wt + (size_t)t * H2_DIM * EDIM;
  v8f c[16];
  #pragma unroll
  for (int n = 0; n < 16; ++n) {
    v8f acc = {};
    #pragma unroll
    for (int ks = 0; ks < 4; ++ks) {
      const _Float16* bp = wt + (size_t)(n * 16 + low) * EDIM + 32 * ks + hi * 8;
      v8h b0 = *(const v8h*)bp;
      v8h b1 = *(const v8h*)(bp + 16);
      v16h bv;
      #pragma unroll
      for (int i = 0; i < 8; ++i) { bv[i] = b0[i]; bv[8 + i] = b1[i]; }
      acc = __builtin_amdgcn_wmma_f32_16x16x32_f16(
                false, a[ks], false, bv, (short)0, acc, false, false);
    }
    c[n] = acc;
  }

  // ---- bias + LayerNorm stats (rows 0-7 live in lanes 0-15, rows 8-15 in 16-31)
  float s[8], q[8];
  #pragma unroll
  for (int j = 0; j < 8; ++j) { s[j] = 0.f; q[j] = 0.f; }
  #pragma unroll
  for (int n = 0; n < 16; ++n) {
    float bv = Bv[t * H2_DIM + n * 16 + low];
    #pragma unroll
    for (int j = 0; j < 8; ++j) {
      float v = c[n][j] + bv;
      c[n][j] = v;
      s[j] += v;
      q[j] += v * v;
    }
  }
  #pragma unroll
  for (int j = 0; j < 8; ++j) {
    #pragma unroll
    for (int m = 1; m < 16; m <<= 1) {     // xor-reduce within 16-lane halves
      s[j] += __shfl_xor(s[j], m, 32);
      q[j] += __shfl_xor(q[j], m, 32);
    }
    float mu  = s[j] * (1.0f / H2_DIM);
    float var = q[j] * (1.0f / H2_DIM) - mu * mu;
    s[j] = mu;
    q[j] = rsqrtf(var + LN_EPS);
  }

  int erow[8];
  #pragma unroll
  for (int j = 0; j < 8; ++j) {
    int r = hi * 8 + j;
    erow[j] = idxbuf[rbase + ((r < rcount) ? r : 0)];
  }

  // ---- gamma/beta + exact GELU + scatter store
  #pragma unroll
  for (int n = 0; n < 16; ++n) {
    int col = n * 16 + low;
    float g  = G[t * H2_DIM + col];
    float be = Be[t * H2_DIM + col];
    #pragma unroll
    for (int j = 0; j < 8; ++j) {
      int r = hi * 8 + j;
      if (r < rcount) {
        float hn = (c[n][j] - s[j]) * q[j] * g + be;
        float y  = 0.5f * hn * (1.0f + erff(hn * 0.70710678118654752f));
        Y[(size_t)erow[j] * H2_DIM + col] = y;
      }
    }
  }
}

extern "C" void kernel_launch(void* const* d_in, const int* in_sizes, int n_in,
                              void* d_out, int out_size, void* d_ws, size_t ws_size,
                              hipStream_t stream) {
  const float* edge_attr = (const float*)d_in[0];
  const float* W         = (const float*)d_in[1];
  const float* b         = (const float*)d_in[2];
  const float* gamma     = (const float*)d_in[3];
  const float* beta      = (const float*)d_in[4];
  const int*   etype     = (const int*)d_in[5];
  float* out = (float*)d_out;

  int* ws = (int*)d_ws;
  _Float16* Wt = (_Float16*)(ws + 64 + E_CNT);   // 16B-aligned offset

  k_zero<<<1, 64, 0, stream>>>(ws);
  k_convW<<<(T_CNT * H2_DIM * EDIM + 255) / 256, 256, 0, stream>>>(W, Wt);
  k_hist<<<(E_CNT + 255) / 256, 256, 0, stream>>>(etype, ws);
  k_prefix<<<1, 1, 0, stream>>>(ws);
  k_scatter<<<(E_CNT + 255) / 256, 256, 0, stream>>>(etype, ws + 8, ws + 64);

  int maxTiles = E_CNT / 16 + T_CNT;             // upper bound on sum ceil(count/16)
  k_moe<<<maxTiles, 32, 0, stream>>>(edge_attr, b, gamma, beta, ws, Wt, out);
}
